// CausalSelfAttention_59399397704253
// MI455X (gfx1250) — matrix operand order repaired and checked by reading, not compiled, not run on hardware
//
#include <hip/hip_runtime.h>

#define TT 2048
#define CC 1024
#define HH 16
#define DH 64
#define BB 4
#define MM (BB * TT) // 8192

typedef __bf16 bf16;
typedef __attribute__((ext_vector_type(16))) __bf16 v16bf;
typedef __attribute__((ext_vector_type(8)))  __bf16 v8bf;
typedef __attribute__((ext_vector_type(4)))  __bf16 v4bf;
typedef __attribute__((ext_vector_type(8)))  float  v8f;
typedef __attribute__((ext_vector_type(4)))  float  v4f;
typedef __attribute__((ext_vector_type(4)))  int    v4i;
// AS-qualified pointee types for the async-to-LDS builtin:
typedef __attribute__((address_space(1))) v4i* gptr_v4i; // global
typedef __attribute__((address_space(3))) v4i* lptr_v4i; // LDS

#define WMMA_BF16(a, b, c) \
  __builtin_amdgcn_wmma_f32_16x16x32_bf16(false, (a), false, (b), (short)0, (c), false, false)

#if defined(__has_builtin)
#  if __has_builtin(__builtin_amdgcn_global_load_async_to_lds_b128)
#    define HAVE_ASYNC_LDS 1
#  endif
#endif
#ifndef HAVE_ASYNC_LDS
#  define HAVE_ASYNC_LDS 0
#endif

static __device__ __forceinline__ v16bf pack16(v8bf lo, v8bf hi) {
  v16bf f;
#pragma unroll
  for (int i = 0; i < 8; ++i) { f[i] = lo[i]; f[i + 8] = hi[i]; }
  return f;
}

// A-fragment (16xK tile of 16-bit data), ISA 7.12.2:
// lanes 0-15: M=lane, K in {0..7, 16..23}; lanes 16-31: M=lane-16, K in {8..15, 24..31}
static __device__ __forceinline__ v16bf load_a_frag(const bf16* base, int stride, int lane) {
  const int r  = lane & 15;
  const int h8 = (lane >> 4) << 3;
  const bf16* p = base + r * stride;
  v8bf lo = *(const v8bf*)(p + h8);
  v8bf hi = *(const v8bf*)(p + 16 + h8);
  return pack16(lo, hi);
}

// B-fragment (32x16, 16-bit): lane n (0-15) holds column n, K in {0..7, 16..23}; lane n+16 holds K in {8..15, 24..31}.
// `base` points at row-major [N][K] storage (i.e. B transposed).
static __device__ __forceinline__ v16bf load_b_frag(const bf16* base, int stride, int lane) {
  const int n  = lane & 15;
  const int k0 = (lane >> 4) << 3;
  const bf16* p = base + n * stride + k0;
  v8bf lo = *(const v8bf*)(p);
  v8bf hi = *(const v8bf*)(p + 16);
  return pack16(lo, hi);
}

// Stage a 128x32 bf16 tile (row stride CC in global) into LDS, 16B chunks per thread.
static __device__ __forceinline__ void stage_tile(bf16* lds_dst, const bf16* gsrc, int tid) {
#pragma unroll
  for (int rep = 0; rep < 2; ++rep) {
    const int c = tid + rep * 256; // 512 chunks of 16B
    const int r = c >> 2;
    const int col = (c & 3) * 8;
    const bf16* src = gsrc + (size_t)r * CC + col;
    bf16* dst = lds_dst + c * 8;
#if HAVE_ASYNC_LDS
    __builtin_amdgcn_global_load_async_to_lds_b128(
        (gptr_v4i)(void*)src, (lptr_v4i)(void*)dst, 0, 0);
#else
    *(v8bf*)dst = *(const v8bf*)src;
#endif
  }
}

static __device__ __forceinline__ void wait_async_zero() {
#if HAVE_ASYNC_LDS
#  if defined(__has_builtin) && __has_builtin(__builtin_amdgcn_s_wait_asynccnt)
  __builtin_amdgcn_s_wait_asynccnt(0);
#  else
  asm volatile("s_wait_asynccnt 0x0" ::: "memory");
#  endif
#endif
}

// Double-buffered LDS-staged 128x128 GEMM mainloop (8 waves: 4x2 16x16 tiles per wave).
static __device__ __forceinline__ void gemm_mainloop(
    const bf16* __restrict__ Ab, const bf16* __restrict__ Bb,
    bf16* ldsA0, bf16* ldsA1, bf16* ldsB0, bf16* ldsB1,
    int tid, int lane, int wr, int wc, v8f acc[4][2]) {
  stage_tile(ldsA0, Ab, tid);
  stage_tile(ldsB0, Bb, tid);
  for (int it = 0; it < CC / 32; ++it) {
    bf16* la = (it & 1) ? ldsA1 : ldsA0;
    bf16* lb = (it & 1) ? ldsB1 : ldsB0;
    wait_async_zero();
    __syncthreads();
    if (it + 1 < CC / 32) {
      bf16* na = (it & 1) ? ldsA0 : ldsA1;
      bf16* nb = (it & 1) ? ldsB0 : ldsB1;
      stage_tile(na, Ab + (it + 1) * 32, tid);
      stage_tile(nb, Bb + (it + 1) * 32, tid);
    }
    v16bf a[4], b[2];
#pragma unroll
    for (int i = 0; i < 4; ++i) a[i] = load_a_frag(la + (wr + 16 * i) * 32, 32, lane);
#pragma unroll
    for (int j = 0; j < 2; ++j) b[j] = load_b_frag(lb + (wc + 16 * j) * 32, 32, lane);
#pragma unroll
    for (int i = 0; i < 4; ++i)
#pragma unroll
      for (int j = 0; j < 2; ++j) acc[i][j] = WMMA_BF16(a[i], b[j], acc[i][j]);
    __syncthreads();
  }
}

// ---------------- f32 -> bf16 conversion ----------------
__global__ void cvt_f32_bf16(const v4f* __restrict__ src, v4bf* __restrict__ dst, int n4) {
  int i = blockIdx.x * blockDim.x + threadIdx.x;
  if (i < n4) {
    v4f f = src[i];
    v4bf o;
#pragma unroll
    for (int j = 0; j < 4; ++j) o[j] = (bf16)f[j];
    dst[i] = o;
  }
}

// ---------------- fused QKV projection GEMM ----------------
// out[m,n] = sum_k x[m,k] * W[n,k];  Q,K stored [B,H,T,64]; V stored transposed [B,H,64,T]
__global__ __launch_bounds__(256) void qkv_gemm(
    const bf16* __restrict__ xb,
    const bf16* __restrict__ wq, const bf16* __restrict__ wk, const bf16* __restrict__ wv,
    bf16* __restrict__ qo, bf16* __restrict__ ko, bf16* __restrict__ vto) {
  __shared__ __align__(16) bf16 ldsA[2][128 * 32];
  __shared__ __align__(16) bf16 ldsB[2][128 * 32];
  const int tid = threadIdx.x, lane = tid & 31, wid = tid >> 5;
  const int wr = (wid & 1) * 64, wc = (wid >> 1) * 32;
  const int row0 = blockIdx.x * 128, col0 = blockIdx.y * 128;
  const int which = blockIdx.z;
  const bf16* w = (which == 0) ? wq : (which == 1) ? wk : wv;

  v8f acc[4][2];
#pragma unroll
  for (int i = 0; i < 4; ++i)
#pragma unroll
    for (int j = 0; j < 2; ++j) acc[i][j] = (v8f){};

  gemm_mainloop(xb + (size_t)row0 * CC, w + (size_t)col0 * CC,
                ldsA[0], ldsA[1], ldsB[0], ldsB[1], tid, lane, wr, wc, acc);

  const int hh = lane >> 4, nn = lane & 15;
  if (which == 2) {
#pragma unroll
    for (int i = 0; i < 4; ++i)
#pragma unroll
      for (int j = 0; j < 2; ++j)
#pragma unroll
        for (int v = 0; v < 8; ++v) {
          int m = row0 + wr + 16 * i + v + 8 * hh;
          int n = col0 + wc + 16 * j + nn;
          int bidx = m >> 11, t = m & (TT - 1);
          int h = n >> 6, d = n & 63;
          vto[((size_t)(bidx * HH + h) * DH + d) * TT + t] = (bf16)acc[i][j][v];
        }
  } else {
    bf16* dst = (which == 1) ? ko : qo;
#pragma unroll
    for (int i = 0; i < 4; ++i)
#pragma unroll
      for (int j = 0; j < 2; ++j)
#pragma unroll
        for (int v = 0; v < 8; ++v) {
          int m = row0 + wr + 16 * i + v + 8 * hh;
          int n = col0 + wc + 16 * j + nn;
          int bidx = m >> 11, t = m & (TT - 1);
          int h = n >> 6, d = n & 63;
          dst[((size_t)(bidx * HH + h) * TT + t) * DH + d] = (bf16)acc[i][j][v];
        }
  }
}

// ---------------- flash-style causal attention ----------------
// grid: (T/128, B*H); 8 waves, each owns 16 query rows.
__global__ __launch_bounds__(256) void attn_kernel(
    const bf16* __restrict__ q, const bf16* __restrict__ k,
    const bf16* __restrict__ vt, bf16* __restrict__ yb) {
  __shared__ __align__(16) bf16 pl[8][16 * 32];
  const int lane = threadIdx.x & 31;
  const int wid  = threadIdx.x >> 5;
  const int bh = blockIdx.y;
  const int q0 = blockIdx.x * 128 + wid * 16;
  const bf16* qbh = q  + (size_t)bh * TT * DH;
  const bf16* kbh = k  + (size_t)bh * TT * DH;
  const bf16* vbh = vt + (size_t)bh * DH * TT;

  const v16bf qf0 = load_a_frag(qbh + (size_t)q0 * DH + 0,  DH, lane);
  const v16bf qf1 = load_a_frag(qbh + (size_t)q0 * DH + 32, DH, lane);

  v8f o[4];
#pragma unroll
  for (int t = 0; t < 4; ++t) o[t] = (v8f){};
  float mi[8], li[8];
#pragma unroll
  for (int v = 0; v < 8; ++v) { mi[v] = -3.0e38f; li[v] = 0.0f; }

  const int hh = lane >> 4, nn = lane & 15;
  bf16* pb = &pl[wid][0];

  for (int j0 = 0; j0 < q0 + 16; j0 += 32) {
    v16bf kf00 = load_b_frag(kbh + (size_t)j0 * DH + 0,         DH, lane);
    v16bf kf01 = load_b_frag(kbh + (size_t)j0 * DH + 32,        DH, lane);
    v16bf kf10 = load_b_frag(kbh + (size_t)(j0 + 16) * DH + 0,  DH, lane);
    v16bf kf11 = load_b_frag(kbh + (size_t)(j0 + 16) * DH + 32, DH, lane);
    v8f s0 = (v8f){}, s1 = (v8f){};
    s0 = WMMA_BF16(qf0, kf00, s0);
    s0 = WMMA_BF16(qf1, kf01, s0);
    s1 = WMMA_BF16(qf0, kf10, s1);
    s1 = WMMA_BF16(qf1, kf11, s1);

    float p0[8], p1[8], al[8];
#pragma unroll
    for (int v = 0; v < 8; ++v) {
      int qg = q0 + v + 8 * hh;
      p0[v] = (j0 + nn      <= qg) ? s0[v] * 0.125f : -3.0e38f;
      p1[v] = (j0 + 16 + nn <= qg) ? s1[v] * 0.125f : -3.0e38f;
    }
#pragma unroll
    for (int v = 0; v < 8; ++v) {
      float rm = fmaxf(p0[v], p1[v]);
#pragma unroll
      for (int off = 8; off; off >>= 1) rm = fmaxf(rm, __shfl_xor(rm, off, 32));
      float mnew = fmaxf(mi[v], rm);
      al[v] = __expf(mi[v] - mnew);
      mi[v] = mnew;
      p0[v] = __expf(p0[v] - mnew);
      p1[v] = __expf(p1[v] - mnew);
      float rs = p0[v] + p1[v];
#pragma unroll
      for (int off = 8; off; off >>= 1) rs += __shfl_xor(rs, off, 32);
      li[v] = li[v] * al[v] + rs;
    }
#pragma unroll
    for (int t = 0; t < 4; ++t)
#pragma unroll
      for (int v = 0; v < 8; ++v) o[t][v] *= al[v];

    // C-layout P -> LDS (row-major 16x32 bf16), then reload in A-fragment layout.
#pragma unroll
    for (int v = 0; v < 8; ++v) {
      int mm = v + 8 * hh;
      pb[mm * 32 + nn]      = (bf16)p0[v];
      pb[mm * 32 + 16 + nn] = (bf16)p1[v];
    }
    // same-wave DS ops are in-order; no cross-wave sharing, so no barrier needed
    {
      const int r = lane & 15, h8 = hh << 3;
      v8bf plo = *(const v8bf*)(pb + r * 32 + h8);
      v8bf phi = *(const v8bf*)(pb + r * 32 + 16 + h8);
      v16bf pf = pack16(plo, phi);
#pragma unroll
      for (int t = 0; t < 4; ++t) {
        v16bf vf = load_b_frag(vbh + (size_t)(t * 16) * TT + j0, TT, lane);
        o[t] = WMMA_BF16(pf, vf, o[t]);
      }
    }
  }

  const int bidx = bh >> 4, h = bh & 15;
#pragma unroll
  for (int v = 0; v < 8; ++v) li[v] = 1.0f / li[v];
#pragma unroll
  for (int t = 0; t < 4; ++t)
#pragma unroll
    for (int v = 0; v < 8; ++v) {
      int mm = v + 8 * hh;
      int tg = q0 + mm;
      int c  = h * 64 + t * 16 + nn;
      yb[(size_t)(bidx * TT + tg) * CC + c] = (bf16)(o[t][v] * li[v]);
    }
}

// ---------------- output projection GEMM with bias ----------------
__global__ __launch_bounds__(256) void out_gemm(
    const bf16* __restrict__ yb, const bf16* __restrict__ wo,
    const float* __restrict__ bo, float* __restrict__ out) {
  __shared__ __align__(16) bf16 ldsA[2][128 * 32];
  __shared__ __align__(16) bf16 ldsB[2][128 * 32];
  const int tid = threadIdx.x, lane = tid & 31, wid = tid >> 5;
  const int wr = (wid & 1) * 64, wc = (wid >> 1) * 32;
  const int row0 = blockIdx.x * 128, col0 = blockIdx.y * 128;

  v8f acc[4][2];
#pragma unroll
  for (int i = 0; i < 4; ++i)
#pragma unroll
    for (int j = 0; j < 2; ++j) acc[i][j] = (v8f){};

  gemm_mainloop(yb + (size_t)row0 * CC, wo + (size_t)col0 * CC,
                ldsA[0], ldsA[1], ldsB[0], ldsB[1], tid, lane, wr, wc, acc);

  const int hh = lane >> 4, nn = lane & 15;
#pragma unroll
  for (int i = 0; i < 4; ++i)
#pragma unroll
    for (int j = 0; j < 2; ++j)
#pragma unroll
      for (int v = 0; v < 8; ++v) {
        int m = row0 + wr + 16 * i + v + 8 * hh;
        int n = col0 + wc + 16 * j + nn;
        out[(size_t)m * CC + n] = acc[i][j][v] + bo[n];
      }
}

extern "C" void kernel_launch(void* const* d_in, const int* in_sizes, int n_in,
                              void* d_out, int out_size, void* d_ws, size_t ws_size,
                              hipStream_t stream) {
  (void)in_sizes; (void)n_in; (void)out_size; (void)ws_size;
  const float* x  = (const float*)d_in[0];
  const float* Wq = (const float*)d_in[1];
  const float* Wk = (const float*)d_in[2];
  const float* Wv = (const float*)d_in[3];
  const float* Wo = (const float*)d_in[4];
  const float* bo = (const float*)d_in[5];
  float* out = (float*)d_out;

  char* ws = (char*)d_ws;
  size_t off = 0;
  bf16* xb  = (bf16*)(ws + off); off += (size_t)MM * CC * 2;  // 16 MB
  bf16* wqb = (bf16*)(ws + off); off += (size_t)CC * CC * 2;  //  2 MB
  bf16* wkb = (bf16*)(ws + off); off += (size_t)CC * CC * 2;
  bf16* wvb = (bf16*)(ws + off); off += (size_t)CC * CC * 2;
  bf16* wob = (bf16*)(ws + off); off += (size_t)CC * CC * 2;
  bf16* qw  = (bf16*)(ws + off); off += (size_t)MM * CC * 2;  // [B,H,T,64]
  bf16* kw  = (bf16*)(ws + off); off += (size_t)MM * CC * 2;  // [B,H,T,64]
  bf16* vtw = (bf16*)(ws + off); off += (size_t)MM * CC * 2;  // [B,H,64,T]
  bf16* ybw = (bf16*)(ws + off); off += (size_t)MM * CC * 2;  // [B,T,C]

  int n4 = MM * CC / 4;
  cvt_f32_bf16<<<(n4 + 255) / 256, 256, 0, stream>>>((const v4f*)x, (v4bf*)xb, n4);
  n4 = CC * CC / 4;
  cvt_f32_bf16<<<(n4 + 255) / 256, 256, 0, stream>>>((const v4f*)Wq, (v4bf*)wqb, n4);
  cvt_f32_bf16<<<(n4 + 255) / 256, 256, 0, stream>>>((const v4f*)Wk, (v4bf*)wkb, n4);
  cvt_f32_bf16<<<(n4 + 255) / 256, 256, 0, stream>>>((const v4f*)Wv, (v4bf*)wvb, n4);
  cvt_f32_bf16<<<(n4 + 255) / 256, 256, 0, stream>>>((const v4f*)Wo, (v4bf*)wob, n4);

  qkv_gemm<<<dim3(MM / 128, CC / 128, 3), 256, 0, stream>>>(xb, wqb, wkb, wvb, qw, kw, vtw);
  attn_kernel<<<dim3(TT / 128, BB * HH), 256, 0, stream>>>(qw, kw, vtw, ybw);
  out_gemm<<<dim3(MM / 128, CC / 128), 256, 0, stream>>>(ybw, wob, bo, out);
}
